// aten_lstm_60997125538166
// MI455X (gfx1250) — compile-verified, matrix operand order checked by reading, not run on hardware
//
#include <hip/hip_runtime.h>
#include <hip/hip_bf16.h>
#include <stdint.h>

// ---------------------------------------------------------------------------
// Bidirectional 2-layer LSTM for MI455X (gfx1250), wave32 + WMMA bf16.
//   Phase 1 (per dir/layer): gx = x @ w_ih^T + b_ih + b_hh  (big WMMA GEMM)
//   Phase 2 (per layer):     persistent scan kernel, w_hh resident in LDS,
//                            c-state in registers, global barrier per step.
// ---------------------------------------------------------------------------

typedef __bf16 bf16_t;
typedef bf16_t v16bf __attribute__((ext_vector_type(16)));
typedef float  v8f   __attribute__((ext_vector_type(8)));

#define T_SEQ 512
#define BATCH 64
#define DIM   512
#define HID   512
#define G4H   2048           // 4*H
#define LDSW  520            // LDS weight row stride (bf16), padded vs bank conflicts
#define GBW   264            // LDS gate-buffer row stride (bf16), padded

// ---------------- f32 -> bf16 convert (grid-stride) ----------------
__global__ void k_f32_to_bf16(const float* __restrict__ src,
                              bf16_t* __restrict__ dst, long n) {
  long i = (long)blockIdx.x * blockDim.x + threadIdx.x;
  long stride = (long)gridDim.x * blockDim.x;
  for (; i < n; i += stride) dst[i] = (bf16_t)src[i];
}

// ---------------- 16x32 bf16 fragment load (A layout; B mirrored with lane=N)
// ISA 7.12.2: lanes 0-15 -> rows, k pairs per dword; lanes 16-31 -> +8 k offset,
// VGPRs 4-7 -> +16 k offset.
__device__ __forceinline__ v16bf load_frag(const bf16_t* __restrict__ base,
                                           int stride, int row16, int kt, int lane) {
  int m   = row16 + (lane & 15);
  int kb4 = (lane >> 4) << 2;                       // dword offset 0 or 4
  const uint32_t* p = (const uint32_t*)(base + (long)m * stride + kt * 32);
  union { v16bf v; uint32_t u[8]; } r;
#pragma unroll
  for (int v = 0; v < 8; ++v)
    r.u[v] = p[((v >> 2) << 3) + kb4 + (v & 3)];
  return r.v;
}

// ---------------- Phase 1: gx[r,n] = sum_k A[r,k]*W[n,k] + b_ih[n] + b_hh[n]
// A: [32768,K] bf16 row-major; W: [2048,K] bf16 row-major; out gx bf16 [32768,2048]
// Block tile 64x256, 8 waves, each wave: 4 M-tiles x 2 N-tiles of 16x16.
__global__ __launch_bounds__(256) void k_gemm_gx(
    const bf16_t* __restrict__ A, const bf16_t* __restrict__ W,
    const float* __restrict__ b_ih, const float* __restrict__ b_hh,
    bf16_t* __restrict__ gx, int K) {
  const int lane  = threadIdx.x & 31;
  const int wv    = threadIdx.x >> 5;
  const long arow0 = (long)blockIdx.x * 64;
  const int  ncol0 = blockIdx.y * 256 + wv * 32;

  v8f acc[4][2];
#pragma unroll
  for (int mt = 0; mt < 4; ++mt)
#pragma unroll
    for (int q = 0; q < 2; ++q)
#pragma unroll
      for (int e = 0; e < 8; ++e) acc[mt][q][e] = 0.f;

  const int nk = K >> 5;
  const bf16_t* Ab = A + arow0 * K;
  const bf16_t* Wb = W + (long)ncol0 * K;
  for (int kt = 0; kt < nk; ++kt) {
    v16bf af[4], bf[2];
#pragma unroll
    for (int mt = 0; mt < 4; ++mt) af[mt] = load_frag(Ab, K, mt * 16, kt, lane);
#pragma unroll
    for (int q = 0; q < 2; ++q)    bf[q]  = load_frag(Wb, K, q * 16, kt, lane);
    if (kt + 1 < nk) {             // CDNA5 global_prefetch_b8 hints for next k-slab
      __builtin_prefetch(Ab + (long)(lane & 15) * K + (kt + 1) * 32, 0, 1);
      __builtin_prefetch(Wb + (long)(lane & 15) * K + (kt + 1) * 32, 0, 1);
    }
#pragma unroll
    for (int mt = 0; mt < 4; ++mt)
#pragma unroll
      for (int q = 0; q < 2; ++q)
        acc[mt][q] = __builtin_amdgcn_wmma_f32_16x16x32_bf16(
            false, af[mt], false, bf[q], (short)0, acc[mt][q], false, false);
  }
  // epilogue: bias add, store bf16.  C/D layout: m = e + 8*(lane>=16), n = lane&15
#pragma unroll
  for (int mt = 0; mt < 4; ++mt)
#pragma unroll
    for (int q = 0; q < 2; ++q) {
      int n = ncol0 + q * 16 + (lane & 15);
      float bias = b_ih[n] + b_hh[n];
#pragma unroll
      for (int e = 0; e < 8; ++e) {
        int m = mt * 16 + e + ((lane >> 4) << 3);
        gx[(arow0 + m) * G4H + n] = (bf16_t)(acc[mt][q][e] + bias);
      }
    }
}

// ---------------- Phase 2: persistent recurrent scan, both directions.
// Grid = 16 blocks: blockIdx>>3 = dir (0 fwd, 1 bwd), blockIdx&7 = hidden chunk.
// Each block owns 64 hidden units => 256 gate rows of w_hh resident in LDS.
__global__ __launch_bounds__(256) void k_lstm_scan(
    const bf16_t* __restrict__ gx_f, const bf16_t* __restrict__ gx_b,
    const bf16_t* __restrict__ whh_f, const bf16_t* __restrict__ whh_b,
    bf16_t* __restrict__ hstate,     // [4][64][512] bf16, pre-init from h0
    const float* __restrict__ c0,    // [4][64][512] f32
    int idx_f, int idx_b,
    bf16_t* __restrict__ ybf,        // layer0 output [T*B,1024] bf16 (or null)
    float*  __restrict__ yf32,       // layer1 output -> d_out x region (or null)
    float*  __restrict__ hn, float* __restrict__ cn,   // [4][64][512]
    unsigned* __restrict__ syncc, unsigned nblocks) {
  extern __shared__ bf16_t smem[];
  bf16_t* ldsW = smem;                 // 256 x LDSW  (~266 KB)
  bf16_t* gbuf = smem + 256 * LDSW;    // 64  x GBW   (~34 KB)   total < 320 KB

  const int tid  = threadIdx.x;
  const int lane = tid & 31;
  const int wv   = tid >> 5;
  const int dir  = blockIdx.x >> 3;
  const int hb   = (blockIdx.x & 7) * 64;
  const int idx  = dir ? idx_b : idx_f;

  const bf16_t* gx  = dir ? gx_b : gx_f;
  const bf16_t* whh = dir ? whh_b : whh_f;
  bf16_t* h = hstate + (long)idx * BATCH * HID;

  // Stage this block's 256 w_hh rows (4 gates x 64 units) into LDS once.
  {
    int gate = tid >> 6, j = tid & 63;
    const int4* s4 = (const int4*)(whh + (long)(gate * HID + hb + j) * HID);
    int4* d4 = (int4*)(ldsW + tid * LDSW);     // 1040B row stride, 16B aligned
#pragma unroll 4
    for (int q = 0; q < (HID * 2) / 16; ++q) d4[q] = s4[q];   // ds_store_b128
  }
  __syncthreads();

  // Persistent cell state: 16 (b,j) cells per thread, in registers.
  float creg[16];
#pragma unroll
  for (int e = 0; e < 16; ++e) {
    int cidx = e * 256 + tid;
    int b = cidx >> 6, j = cidx & 63;
    creg[e] = c0[(long)idx * BATCH * HID + b * HID + hb + j];
  }

  for (int s = 0; s < T_SEQ; ++s) {
    const int t = dir ? (T_SEQ - 1 - s) : s;

    v8f acc[4][2];
#pragma unroll
    for (int mt = 0; mt < 4; ++mt)
#pragma unroll
      for (int q = 0; q < 2; ++q)
#pragma unroll
        for (int e = 0; e < 8; ++e) acc[mt][q][e] = 0.f;

    // g_slice[64 x 256] += h[64 x 512] @ Wlds^T[512 x 256]
#pragma unroll 1
    for (int kt = 0; kt < HID / 32; ++kt) {
      v16bf af[4], bf[2];
#pragma unroll
      for (int mt = 0; mt < 4; ++mt) af[mt] = load_frag(h, HID, mt * 16, kt, lane);
#pragma unroll
      for (int q = 0; q < 2; ++q)
        bf[q] = load_frag(ldsW, LDSW, wv * 32 + q * 16, kt, lane);  // ds_load
#pragma unroll
      for (int mt = 0; mt < 4; ++mt)
#pragma unroll
        for (int q = 0; q < 2; ++q)
          acc[mt][q] = __builtin_amdgcn_wmma_f32_16x16x32_bf16(
              false, af[mt], false, bf[q], (short)0, acc[mt][q], false, false);
    }

    // Scatter fragments to LDS gate buffer for cross-wave gate combination.
#pragma unroll
    for (int mt = 0; mt < 4; ++mt)
#pragma unroll
      for (int q = 0; q < 2; ++q) {
        int n = wv * 32 + q * 16 + (lane & 15);
#pragma unroll
        for (int e = 0; e < 8; ++e) {
          int m = mt * 16 + e + ((lane >> 4) << 3);
          gbuf[m * GBW + n] = (bf16_t)acc[mt][q][e];
        }
      }
    __syncthreads();

    // Gates + state update (PyTorch order i,f,g,o).
#pragma unroll
    for (int e = 0; e < 16; ++e) {
      int cidx = e * 256 + tid;
      int b = cidx >> 6, j = cidx & 63;
      long gr = ((long)t * BATCH + b) * G4H + hb + j;
      float gi = (float)gbuf[b * GBW + j]       + (float)gx[gr];
      float gf = (float)gbuf[b * GBW + 64 + j]  + (float)gx[gr + 512];
      float gg = (float)gbuf[b * GBW + 128 + j] + (float)gx[gr + 1024];
      float go = (float)gbuf[b * GBW + 192 + j] + (float)gx[gr + 1536];
      float i_ = 1.f / (1.f + __expf(-gi));
      float f_ = 1.f / (1.f + __expf(-gf));
      float g_ = tanhf(gg);
      float o_ = 1.f / (1.f + __expf(-go));
      float c  = f_ * creg[e] + i_ * g_;
      creg[e]  = c;
      float hv = o_ * tanhf(c);
      h[b * HID + hb + j] = (bf16_t)hv;                 // feeds next step GEMM
      long yoff = ((long)t * BATCH + b) * (2 * HID) + dir * HID + hb + j;
      if (yf32) yf32[yoff] = hv; else ybf[yoff] = (bf16_t)hv;
      if (s == T_SEQ - 1) {
        long so = (long)idx * BATCH * HID + b * HID + hb + j;
        hn[so] = hv; cn[so] = c;
      }
    }
    __syncthreads();

    // Device-wide step barrier: publish h, then wait for all 16 blocks.
    if (s + 1 < T_SEQ) {
      __threadfence();
      if (tid == 0) {
        __hip_atomic_fetch_add(syncc, 1u, __ATOMIC_RELEASE, __HIP_MEMORY_SCOPE_AGENT);
        unsigned target = nblocks * (unsigned)(s + 1);
        while (__hip_atomic_load(syncc, __ATOMIC_ACQUIRE, __HIP_MEMORY_SCOPE_AGENT) < target)
          __builtin_amdgcn_s_sleep(1);
      }
      __syncthreads();
    }
  }
}

// ---------------------------------------------------------------------------
extern "C" void kernel_launch(void* const* d_in, const int* in_sizes, int n_in,
                              void* d_out, int out_size, void* d_ws, size_t ws_size,
                              hipStream_t stream) {
  (void)in_sizes; (void)n_in; (void)out_size; (void)ws_size;
  const float* x  = (const float*)d_in[0];
  const float* h0 = (const float*)d_in[1];
  const float* c0 = (const float*)d_in[2];
  // params idx = 2*layer + dir:  l0 fwd(3..6), l0 rev(7..10), l1 fwd(11..14), l1 rev(15..18)
  const float* w_ih[4] = {(const float*)d_in[3],  (const float*)d_in[7],
                          (const float*)d_in[11], (const float*)d_in[15]};
  const float* w_hh[4] = {(const float*)d_in[4],  (const float*)d_in[8],
                          (const float*)d_in[12], (const float*)d_in[16]};
  const float* b_ih[4] = {(const float*)d_in[5],  (const float*)d_in[9],
                          (const float*)d_in[13], (const float*)d_in[17]};
  const float* b_hh[4] = {(const float*)d_in[6],  (const float*)d_in[10],
                          (const float*)d_in[14], (const float*)d_in[18]};

  char* ws = (char*)d_ws;
  size_t off = 0;
  auto carve = [&](size_t bytes) -> char* {
    char* p = ws + off;
    off = (off + bytes + 255) & ~(size_t)255;
    return p;
  };
  bf16_t* xbf = (bf16_t*)carve((size_t)T_SEQ * BATCH * DIM * 2);       //  32 MB
  bf16_t* ybf = (bf16_t*)carve((size_t)T_SEQ * BATCH * 2 * HID * 2);   //  64 MB
  bf16_t* gx0 = (bf16_t*)carve((size_t)T_SEQ * BATCH * G4H * 2);       // 128 MB
  bf16_t* gx1 = (bf16_t*)carve((size_t)T_SEQ * BATCH * G4H * 2);       // 128 MB
  bf16_t* wihb[4]; bf16_t* whhb[4];
  for (int i = 0; i < 4; ++i) {
    size_t din = (i < 2) ? DIM : 2 * HID;
    wihb[i] = (bf16_t*)carve((size_t)G4H * din * 2);
    whhb[i] = (bf16_t*)carve((size_t)G4H * HID * 2);
  }
  bf16_t*   hbf   = (bf16_t*)carve((size_t)4 * BATCH * HID * 2);
  unsigned* syncp = (unsigned*)carve(512);

  // --- converts (f32 -> bf16): inputs, weights, initial h ---
  const int CB = 256, CG = 4096;
  k_f32_to_bf16<<<CG, CB, 0, stream>>>(x, xbf, (long)T_SEQ * BATCH * DIM);
  k_f32_to_bf16<<<CG, CB, 0, stream>>>(h0, hbf, (long)4 * BATCH * HID);
  for (int i = 0; i < 4; ++i) {
    long din = (i < 2) ? DIM : 2 * HID;
    k_f32_to_bf16<<<CG, CB, 0, stream>>>(w_ih[i], wihb[i], (long)G4H * din);
    k_f32_to_bf16<<<CG, CB, 0, stream>>>(w_hh[i], whhb[i], (long)G4H * HID);
  }
  hipMemsetAsync(syncp, 0, 512, stream);   // reset both step-barrier counters

  float* out = (float*)d_out;
  float* hn  = out + (size_t)T_SEQ * BATCH * 2 * HID;
  float* cn  = hn + (size_t)4 * BATCH * HID;

  size_t shbytes = (size_t)(256 * LDSW + 64 * GBW) * sizeof(bf16_t);
  hipFuncSetAttribute(reinterpret_cast<const void*>(k_lstm_scan),
                      hipFuncAttributeMaxDynamicSharedMemorySize, (int)shbytes);

  dim3 gblk(T_SEQ * BATCH / 64, G4H / 256);

  // ---- layer 0 ----
  k_gemm_gx<<<gblk, 256, 0, stream>>>(xbf, wihb[0], b_ih[0], b_hh[0], gx0, DIM);
  k_gemm_gx<<<gblk, 256, 0, stream>>>(xbf, wihb[1], b_ih[1], b_hh[1], gx1, DIM);
  k_lstm_scan<<<16, 256, shbytes, stream>>>(gx0, gx1, whhb[0], whhb[1], hbf, c0,
                                            0, 1, ybf, nullptr, hn, cn, syncp, 16u);
  // ---- layer 1 ----
  k_gemm_gx<<<gblk, 256, 0, stream>>>(ybf, wihb[2], b_ih[2], b_hh[2], gx0, 2 * HID);
  k_gemm_gx<<<gblk, 256, 0, stream>>>(ybf, wihb[3], b_ih[3], b_hh[3], gx1, 2 * HID);
  k_lstm_scan<<<16, 256, shbytes, stream>>>(gx0, gx1, whhb[2], whhb[3], hbf, c0,
                                            2, 3, nullptr, out, hn, cn, syncp + 64, 16u);
}
